// Agent_Attention_3D_22239340659251
// MI455X (gfx1250) — compile-verified, matrix operand order checked by reading, not run on hardware
//
#include <hip/hip_runtime.h>
#include <hip/hip_bf16.h>
#include <math.h>

// ---------------------------------------------------------------------------
// Types for CDNA5 WMMA (wave32): 16x16x32 bf16 -> f32
// ---------------------------------------------------------------------------
typedef __attribute__((ext_vector_type(16))) __bf16 bf16x16;
typedef __attribute__((ext_vector_type(8)))  float  fx8;
typedef __attribute__((ext_vector_type(4)))  int    i32x4;

#define DEVI __device__ __forceinline__

union FragU { bf16x16 v; uint4 q[2]; };

// ---- Optional CDNA5 async global->LDS staging (ASYNCcnt-tracked) ----------
#if defined(__has_builtin)
#if __has_builtin(__builtin_amdgcn_global_load_async_to_lds_b128)
#define HAS_ASYNC_LDS 1
#endif
#endif

#ifdef HAS_ASYNC_LDS
// Signature (from hipcc diagnostics): (int4 addrspace(1)*, int4 addrspace(3)*,
// imm offset, imm cpol) -- AS(1) prints as "__device__" in HIP diags.
typedef __attribute__((address_space(1))) i32x4 GVec;
typedef __attribute__((address_space(3))) i32x4 LVec;
DEVI void async_copy_b128(const void* g, void* l) {
  __builtin_amdgcn_global_load_async_to_lds_b128((GVec*)g, (LVec*)l, 0, 0);
}
DEVI void wait_async0() {
#if __has_builtin(__builtin_amdgcn_s_wait_asynccnt)
  __builtin_amdgcn_s_wait_asynccnt(0);
#else
  asm volatile("s_wait_asynccnt 0x0" ::: "memory");
#endif
}
#endif

DEVI fx8 wmma_bf16(bf16x16 a, bf16x16 b, fx8 c) {
  // (neg_a, A, neg_b, B, c_mod, C, reuse_a, reuse_b)
  return __builtin_amdgcn_wmma_f32_16x16x32_bf16(false, a, false, b, (short)0, c,
                                                 false, false);
}

DEVI bf16x16 zero_frag() {
  FragU u;
  u.q[0] = make_uint4(0u, 0u, 0u, 0u);
  u.q[1] = make_uint4(0u, 0u, 0u, 0u);
  return u.v;
}

// Load a 16x32 bf16 operand fragment from a row-major [16][ldk] tile.
// ISA layout (16-bit A, 16x32): lanes 0-15 row=lane, K 0..7 in V0-3 and
// K 16..23 in V4-7; lanes 16-31 row=lane-16, K 8..15 and K 24..31.
// => per lane: two 16B loads at (row, kb) and (row, kb+16), kb = (lane>=16)*8.
// B operand (32x16, stored transposed as [N][K]) uses the same pattern.
DEVI bf16x16 load_frag(const __bf16* tile, int ldk) {
  const int lane = threadIdx.x & 31;
  const int r  = lane & 15;
  const int kb = (lane >> 4) * 8;
  const __bf16* p = tile + r * ldk + kb;
  FragU u;
  u.q[0] = *reinterpret_cast<const uint4*>(p);
  u.q[1] = *reinterpret_cast<const uint4*>(p + 16);
  return u.v;
}

// ---------------------------------------------------------------------------
// Dense GEMM: C[M][N] = A[M][K](bf16,rowmajor) x W[N][K](bf16,rowmajor)^T + bias
// Block tile 128x64, 8 waves (4x2), each wave 32x32 via 2x2 WMMA frags, K step 32.
// Staging uses the CDNA5 async global->LDS path when available.
// ---------------------------------------------------------------------------
__global__ __launch_bounds__(256)
void gemm_bf16_kernel(const __bf16* __restrict__ A, const __bf16* __restrict__ W,
                      const float* __restrict__ bias, float* __restrict__ C,
                      int M, int N, int K) {
  const int LDK = 40;                     // 32 + 8 pad (80B rows, conflict-free b128)
  __shared__ __bf16 sA[128 * 40];
  __shared__ __bf16 sB[64 * 40];

  const int wave = threadIdx.x >> 5;
  const int lane = threadIdx.x & 31;
  const int lh = lane >> 4, ln = lane & 15;
  const int m0 = blockIdx.y * 128;
  const int n0 = blockIdx.x * 64;
  const int wm = (wave & 3) * 32;
  const int wn = (wave >> 2) * 32;

  fx8 acc00 = {0.f,0.f,0.f,0.f,0.f,0.f,0.f,0.f};
  fx8 acc01 = acc00, acc10 = acc00, acc11 = acc00;

  for (int k0 = 0; k0 < K; k0 += 32) {
    // Stage A tile (128 rows x 64B): 512 x 16B chunks; B tile: 256 x 16B chunks.
#ifdef HAS_ASYNC_LDS
#pragma unroll
    for (int i = 0; i < 2; ++i) {
      const int idx = threadIdx.x + i * 256;
      const int r = idx >> 2, c = (idx & 3) * 8;
      async_copy_b128(&A[(size_t)(m0 + r) * K + k0 + c], &sA[r * LDK + c]);
    }
    {
      const int r = threadIdx.x >> 2, c = (threadIdx.x & 3) * 8;
      async_copy_b128(&W[(size_t)(n0 + r) * K + k0 + c], &sB[r * LDK + c]);
    }
#else
#pragma unroll
    for (int i = 0; i < 2; ++i) {
      const int idx = threadIdx.x + i * 256;
      const int r = idx >> 2, c = (idx & 3) * 8;
      *reinterpret_cast<uint4*>(&sA[r * LDK + c]) =
          *reinterpret_cast<const uint4*>(&A[(size_t)(m0 + r) * K + k0 + c]);
    }
    {
      const int r = threadIdx.x >> 2, c = (threadIdx.x & 3) * 8;
      *reinterpret_cast<uint4*>(&sB[r * LDK + c]) =
          *reinterpret_cast<const uint4*>(&W[(size_t)(n0 + r) * K + k0 + c]);
    }
#endif
    // Prefetch next K tile into cache (global_prefetch_b8).
    if (k0 + 32 < K) {
      __builtin_prefetch(&A[(size_t)(m0 + (threadIdx.x >> 1)) * K + k0 + 32], 0, 1);
      if (threadIdx.x < 64)
        __builtin_prefetch(&W[(size_t)(n0 + threadIdx.x) * K + k0 + 32], 0, 1);
    }
#ifdef HAS_ASYNC_LDS
    wait_async0();                        // own async copies landed in LDS
#endif
    __syncthreads();                      // everyone's copies landed

    const bf16x16 a0 = load_frag(&sA[(wm)      * LDK], LDK);
    const bf16x16 a1 = load_frag(&sA[(wm + 16) * LDK], LDK);
    const bf16x16 b0 = load_frag(&sB[(wn)      * LDK], LDK);
    const bf16x16 b1 = load_frag(&sB[(wn + 16) * LDK], LDK);

    acc00 = wmma_bf16(a0, b0, acc00);
    acc01 = wmma_bf16(a0, b1, acc01);
    acc10 = wmma_bf16(a1, b0, acc10);
    acc11 = wmma_bf16(a1, b1, acc11);
    __syncthreads();
  }

  // Epilogue. C layout: VGPR r, lanes 0-15: M=r, N=lane; lanes 16-31: M=r+8.
  {
    const int col0 = n0 + wn + ln;
    const int col1 = col0 + 16;
    const float bb0 = bias[col0], bb1 = bias[col1];
#pragma unroll
    for (int r = 0; r < 8; ++r) {
      const int row0 = m0 + wm + lh * 8 + r;
      const int row1 = row0 + 16;
      C[(size_t)row0 * N + col0] = acc00[r] + bb0;
      C[(size_t)row0 * N + col1] = acc01[r] + bb1;
      C[(size_t)row1 * N + col0] = acc10[r] + bb0;
      C[(size_t)row1 * N + col1] = acc11[r] + bb1;
    }
  }
}

// ---------------------------------------------------------------------------
// 3x3x3 conv (full 256->256 channels) as implicit GEMM with WMMA.
// Vnc[n][c] bf16 (NHWC over 32^3), Wt[tap][o][i] bf16. Accumulates into O += .
// Block = 64(M spatial) x 64(N out-ch); 8 waves, each 16x32 (2 WMMA N-frags).
// Both operands are L2-resident (16MB + 3.4MB << 192MB L2) -> direct loads.
// ---------------------------------------------------------------------------
__global__ __launch_bounds__(256)
void conv_wmma_kernel(const __bf16* __restrict__ Vnc, const __bf16* __restrict__ Wt,
                      const float* __restrict__ b_dwc, float* __restrict__ O) {
  const int wave = threadIdx.x >> 5;
  const int lane = threadIdx.x & 31;
  const int lh = lane >> 4, ln = lane & 15;
  const int m0 = blockIdx.y * 64 + (wave >> 1) * 16;
  const int n0 = blockIdx.x * 64 + (wave & 1) * 32;

  // This lane's A-fragment spatial row.
  const int mrow = m0 + ln;
  const int z = mrow >> 10, y = (mrow >> 5) & 31, x = mrow & 31;

  fx8 acc0 = {0.f,0.f,0.f,0.f,0.f,0.f,0.f,0.f};
  fx8 acc1 = acc0;

  for (int t = 0; t < 27; ++t) {
    const int dz = t / 9 - 1;
    const int dy = (t / 3) % 3 - 1;
    const int dx = t % 3 - 1;
    const int zz = z + dz, yy = y + dy, xx = x + dx;
    const bool ok = ((unsigned)zz < 32u) & ((unsigned)yy < 32u) & ((unsigned)xx < 32u);
    const int nt = (zz << 10) + (yy << 5) + xx;
    const __bf16* arow = Vnc + (size_t)nt * 256;
    const __bf16* w0 = Wt + ((size_t)t * 256 + n0 + ln) * 256;
    const __bf16* w1 = w0 + 16 * 256;
#pragma unroll
    for (int kc = 0; kc < 8; ++kc) {
      const int koff = kc * 32 + lh * 8;
      bf16x16 af;
      if (ok) {
        FragU u;
        u.q[0] = *reinterpret_cast<const uint4*>(arow + koff);
        u.q[1] = *reinterpret_cast<const uint4*>(arow + koff + 16);
        af = u.v;
      } else {
        af = zero_frag();                 // SAME-padding: zero contribution
      }
      FragU ub0, ub1;
      ub0.q[0] = *reinterpret_cast<const uint4*>(w0 + koff);
      ub0.q[1] = *reinterpret_cast<const uint4*>(w0 + koff + 16);
      ub1.q[0] = *reinterpret_cast<const uint4*>(w1 + koff);
      ub1.q[1] = *reinterpret_cast<const uint4*>(w1 + koff + 16);
      acc0 = wmma_bf16(af, ub0.v, acc0);
      acc1 = wmma_bf16(af, ub1.v, acc1);
    }
  }

  {
    const int col0 = n0 + ln, col1 = col0 + 16;
    const float bb0 = b_dwc[col0], bb1 = b_dwc[col1];
#pragma unroll
    for (int r = 0; r < 8; ++r) {
      const int row = m0 + lh * 8 + r;
      O[(size_t)row * 256 + col0] += acc0[r] + bb0;
      O[(size_t)row * 256 + col1] += acc1[r] + bb1;
    }
  }
}

// ---------------------------------------------------------------------------
// Agent pooling: A_sc[h][a][d] = scale * mean over 8^3 block of the torch-
// faithful interleaved reshape of q. q element at flat idx i lives at
// qkv[n*768 + h*32 + d] with h=i>>20, n=(i>>5)&32767, d=i&31.
// ---------------------------------------------------------------------------
__global__ __launch_bounds__(256)
void pool_kernel(const float* __restrict__ qkv, float* __restrict__ Asc) {
  const int t = blockIdx.x * 256 + threadIdx.x;     // 16384 = 256 ch * 64 agents
  if (t >= 16384) return;
  const int c = t >> 6, a = t & 63;
  const int p1 = a >> 4, p2 = (a >> 2) & 3, p3 = a & 3;
  float sum = 0.f;
  for (int b1 = 0; b1 < 8; ++b1)
    for (int b2 = 0; b2 < 8; ++b2)
      for (int b3 = 0; b3 < 8; ++b3) {
        const int f1 = p1 * 8 + b1, f2 = p2 * 8 + b2, f3 = p3 * 8 + b3;
        const int idx = f1 * 262144 + f2 * 8192 + f3 * 256 + c;
        const int hh = idx >> 20, nn = (idx >> 5) & 32767, dd = idx & 31;
        sum += qkv[(size_t)nn * 768 + hh * 32 + dd];
      }
  const float scale = 0.17677669529663687f;         // 32^-0.5 (folded once)
  Asc[((c >> 5) * 64 + a) * 32 + (c & 31)] = sum * (1.f / 512.f) * scale;
}

// ---------------------------------------------------------------------------
// Stage 1: agent_v[h][a][:] = softmax_n(A_sc[h][a]·k[h][n]) @ v. One wave per
// (head,agent) row; per-lane online softmax over 1024 keys; shfl-xor merge.
// ---------------------------------------------------------------------------
__global__ __launch_bounds__(256)
void stage1_kernel(const float* __restrict__ qkv, const float* __restrict__ Asc,
                   float* __restrict__ agent_v) {
  const int wave = threadIdx.x >> 5, lane = threadIdx.x & 31;
  const int row = blockIdx.x * 8 + wave;            // 0..511
  const int h = row >> 6, a = row & 63;

  float av[32];
#pragma unroll
  for (int d = 0; d < 32; ++d) av[d] = Asc[(h * 64 + a) * 32 + d];

  float m = -1e30f, l = 0.f;
  float acc[32];
#pragma unroll
  for (int d = 0; d < 32; ++d) acc[d] = 0.f;

  const float* kbase = qkv + 256 + h * 32;
  const float* vbase = qkv + 512 + h * 32;
  for (int n = lane; n < 32768; n += 32) {
    const float* kr = kbase + (size_t)n * 768;
    float s = 0.f;
#pragma unroll
    for (int d = 0; d < 32; ++d) s += av[d] * kr[d];
    const float nm = fmaxf(m, s);
    const float corr = __expf(m - nm);
    const float w = __expf(s - nm);
    l = l * corr + w;
    const float* vr = vbase + (size_t)n * 768;
#pragma unroll
    for (int d = 0; d < 32; ++d) acc[d] = acc[d] * corr + w * vr[d];
    m = nm;
  }
  // Wave32 reduction of (m, l, acc[32]) flash-states.
#pragma unroll
  for (int off = 16; off >= 1; off >>= 1) {
    const float m2 = __shfl_xor(m, off, 32);
    const float l2 = __shfl_xor(l, off, 32);
    const float nm = fmaxf(m, m2);
    const float c1 = __expf(m - nm), c2 = __expf(m2 - nm);
    l = l * c1 + l2 * c2;
#pragma unroll
    for (int d = 0; d < 32; ++d)
      acc[d] = acc[d] * c1 + __shfl_xor(acc[d], off, 32) * c2;
    m = nm;
  }
  if (lane == 0) {
    const float inv = 1.f / l;
#pragma unroll
    for (int d = 0; d < 32; ++d)
      agent_v[(h * 64 + a) * 32 + d] = acc[d] * inv;
  }
}

// ---------------------------------------------------------------------------
// Stage 2: per-query softmax over 64 agents, then @ agent_v. Agent tables in
// LDS. Output scattered through the torch-faithful flat index
// h*1048576 + d*32768 + n into the (32768,256) o buffer.
// ---------------------------------------------------------------------------
__global__ __launch_bounds__(256)
void stage2_kernel(const float* __restrict__ qkv, const float* __restrict__ Asc,
                   const float* __restrict__ agent_v, float* __restrict__ O) {
  __shared__ float sA[2048];
  __shared__ float sV[2048];
  const int t = blockIdx.x * 256 + threadIdx.x;
  const int h = t >> 15, n = t & 32767;
  for (int i = threadIdx.x; i < 2048; i += 256) {
    sA[i] = Asc[h * 2048 + i];
    sV[i] = agent_v[h * 2048 + i];
  }
  __syncthreads();

  float q[32];
  const float* qr = qkv + (size_t)n * 768 + h * 32;
#pragma unroll
  for (int d = 0; d < 32; ++d) q[d] = qr[d];

  // Pass 1: row max (scores recomputed in pass 2 to keep registers lean).
  float m = -1e30f;
  for (int a = 0; a < 64; ++a) {
    float s = 0.f;
#pragma unroll
    for (int d = 0; d < 32; ++d) s += q[d] * sA[a * 32 + d];
    m = fmaxf(m, s);
  }
  float l = 0.f;
  float o[32];
#pragma unroll
  for (int d = 0; d < 32; ++d) o[d] = 0.f;
  for (int a = 0; a < 64; ++a) {
    float s = 0.f;
#pragma unroll
    for (int d = 0; d < 32; ++d) s += q[d] * sA[a * 32 + d];
    const float p = __expf(s - m);
    l += p;
#pragma unroll
    for (int d = 0; d < 32; ++d) o[d] += p * sV[a * 32 + d];
  }
  const float inv = 1.f / l;
  const size_t base = (size_t)h * 1048576 + (size_t)n;
#pragma unroll
  for (int d = 0; d < 32; ++d) O[base + (size_t)d * 32768] = o[d] * inv;
}

// ---------------------------------------------------------------------------
// Small conversion / repack kernels.
// ---------------------------------------------------------------------------
__global__ __launch_bounds__(256)
void cvt_bf16_kernel(const float* __restrict__ in, __bf16* __restrict__ out, int n) {
  const int i = blockIdx.x * 256 + threadIdx.x;
  if (i < n) out[i] = (__bf16)in[i];
}

// w_dwc (O=256, I=256, 3,3,3) -> Wt[tap][o][i] bf16
__global__ __launch_bounds__(256)
void repack_dwc_kernel(const float* __restrict__ w, __bf16* __restrict__ out) {
  const int i = blockIdx.x * 256 + threadIdx.x;     // 27*65536 = 1769472
  if (i < 27 * 65536) {
    const int tap = i >> 16, rem = i & 65535, o = rem >> 8, ic = rem & 255;
    out[i] = (__bf16)w[(size_t)o * 6912 + ic * 27 + tap];
  }
}

// qkv_f32 v-slice -> Vnc[n][c] bf16 (NHWC layout for implicit-GEMM conv)
__global__ __launch_bounds__(256)
void repack_vnc_kernel(const float* __restrict__ qkv, __bf16* __restrict__ vnc) {
  const int i = blockIdx.x * 256 + threadIdx.x;     // 32768*256 exactly
  const int n = i >> 8, c = i & 255;
  vnc[i] = (__bf16)qkv[(size_t)n * 768 + 512 + c];
}

// ---------------------------------------------------------------------------
// Host launcher
// ---------------------------------------------------------------------------
extern "C" void kernel_launch(void* const* d_in, const int* in_sizes, int n_in,
                              void* d_out, int out_size, void* d_ws, size_t ws_size,
                              hipStream_t stream) {
  const float* x      = (const float*)d_in[0];   // (32768, 256)
  const float* w_qkv  = (const float*)d_in[1];   // (768, 256)
  const float* b_qkv  = (const float*)d_in[2];   // (768,)
  const float* w_proj = (const float*)d_in[3];   // (256, 256)
  const float* b_proj = (const float*)d_in[4];   // (256,)
  const float* w_dwc  = (const float*)d_in[5];   // (256,256,3,3,3)
  const float* b_dwc  = (const float*)d_in[6];   // (256,)
  float* out = (float*)d_out;                    // (32768, 256)

  char* p = (char*)d_ws;
  auto alloc = [&](size_t bytes) -> char* {
    char* r = p;
    p += (bytes + 255) & ~(size_t)255;
    return r;
  };

  __bf16* xb   = (__bf16*)alloc((size_t)32768 * 256 * 2);     // x bf16
  __bf16* wqb  = (__bf16*)alloc((size_t)768 * 256 * 2);       // w_qkv bf16 [N][K]
  __bf16* wpb  = (__bf16*)alloc((size_t)256 * 256 * 2);       // w_proj bf16 [N][K]
  __bf16* wdb  = (__bf16*)alloc((size_t)27 * 256 * 256 * 2);  // w_dwc repacked
  float*  qkvf = (float*) alloc((size_t)32768 * 768 * 4);     // qkv fp32
  float*  Asc  = (float*) alloc((size_t)8 * 64 * 32 * 4);     // scaled agents
  float*  agv  = (float*) alloc((size_t)8 * 64 * 32 * 4);     // agent_v
  float*  of   = (float*) alloc((size_t)32768 * 256 * 4);     // attn + conv out
  __bf16* ob   = (__bf16*)alloc((size_t)32768 * 256 * 2);     // of bf16
  __bf16* vnc  = (__bf16*)alloc((size_t)32768 * 256 * 2);     // v NHWC bf16
  (void)ws_size; (void)in_sizes; (void)n_in; (void)out_size;

  // 1) Precision conversions / weight repacks.
  cvt_bf16_kernel<<<(8388608 + 255) / 256, 256, 0, stream>>>(x, xb, 8388608);
  cvt_bf16_kernel<<<(196608 + 255) / 256, 256, 0, stream>>>(w_qkv, wqb, 196608);
  cvt_bf16_kernel<<<(65536 + 255) / 256, 256, 0, stream>>>(w_proj, wpb, 65536);
  repack_dwc_kernel<<<1769472 / 256, 256, 0, stream>>>(w_dwc, wdb);

  // 2) QKV GEMM (WMMA): (32768x256)@(256x768) + b_qkv
  gemm_bf16_kernel<<<dim3(768 / 64, 32768 / 128), 256, 0, stream>>>(
      xb, wqb, b_qkv, qkvf, 32768, 768, 256);

  // 3) Agent pooling (scale folded).
  pool_kernel<<<64, 256, 0, stream>>>(qkvf, Asc);

  // 4) Stage 1: agents attend to keys (flash softmax over N=32768).
  stage1_kernel<<<64, 256, 0, stream>>>(qkvf, Asc, agv);

  // 5) Stage 2: queries attend to agents; writes attention output into `of`.
  stage2_kernel<<<1024, 256, 0, stream>>>(qkvf, Asc, agv, of);

  // 6) Conv residual: repack v to NHWC bf16, implicit-GEMM conv += into `of`.
  repack_vnc_kernel<<<32768, 256, 0, stream>>>(qkvf, vnc);
  conv_wmma_kernel<<<dim3(256 / 64, 32768 / 64), 256, 0, stream>>>(
      vnc, wdb, b_dwc, of);

  // 7) Output projection (WMMA): (32768x256)@(256x256) + b_proj -> d_out
  cvt_bf16_kernel<<<32768, 256, 0, stream>>>(of, ob, 8388608);
  gemm_bf16_kernel<<<dim3(256 / 64, 32768 / 128), 256, 0, stream>>>(
      ob, wpb, b_proj, out, 32768, 256, 256);
}
